// LLCMBackbone_11373073400007
// MI455X (gfx1250) — compile-verified, matrix operand order checked by reading, not run on hardware
//
#include <hip/hip_runtime.h>
#include <cstdint>
#include <cstddef>

// ---------------------------------------------------------------------------
// Types for CDNA5 WMMA (wave32): v16bf A/B fragments, v8f f32 accumulator.
// ---------------------------------------------------------------------------
typedef __attribute__((ext_vector_type(16))) __bf16 v16bf;
typedef __attribute__((ext_vector_type(8)))  float  v8f;

struct alignas(16) U128 { unsigned int w[4]; };

// Model constants
#define DIMC   1024
#define TLEN   1024
#define BATCH  4
#define NHEAD  16
#define HDIM   64
#define NTH    4          // "t" heads
#define TDC    256        // NTH*HDIM
#define LAY    4
#define NTOK   (BATCH*TLEN)   // 4096
#define EPSC   1e-5f

// ---------------------------------------------------------------------------
// WMMA helper
// ---------------------------------------------------------------------------
__device__ inline v8f wmma_bf16(v16bf a, v16bf b, v8f c) {
    return __builtin_amdgcn_wmma_f32_16x16x32_bf16(
        /*neg_a=*/false, a, /*neg_b=*/false, b,
        /*c_mod=*/(short)0, c, /*reuse_a=*/false, /*reuse_b=*/false);
}

// CDNA5 async global->LDS copy (16 bytes per lane), tracked by ASYNCcnt.
// LDS address comes from the vdst VGPR plus the instruction offset; global addr from 64-bit VGPR pair.
__device__ inline void async_copy16(unsigned lds_off, const void* gptr) {
    asm volatile("global_load_async_to_lds_b128 %0, %1, off"
                 :: "v"(lds_off), "v"(gptr) : "memory");
}
__device__ inline void wait_asynccnt0() {
    asm volatile("s_wait_asynccnt 0" ::: "memory");
}

// Load a 16x32 bf16 fragment ("A layout": lane r<16 holds row r, K 0..7 & 16..23;
// lanes 16..31 hold K 8..15 & 24..31) from row-major storage with the given
// element stride. For C = A * W^T with W stored (N,K) row-major, the B operand
// is loaded with exactly the same pattern — so one loader serves both.
__device__ inline v16bf load_frag(const __bf16* base, int stride, int lane) {
    int r  = lane & 15;
    int hl = (lane >> 4) & 1;
    const __bf16* p = base + (size_t)r * stride + hl * 8;
    union { v16bf v; U128 q[2]; } u;
    u.q[0] = *(const U128*)p;
    u.q[1] = *(const U128*)(p + 16);
    return u.v;
}

// ---------------------------------------------------------------------------
// Embedding: h[b,t,:] = x[b,t,:] @ Wemb^T + bemb + pos[t,:]
// ---------------------------------------------------------------------------
__global__ __launch_bounds__(256) void embed_kernel(
    const float* __restrict__ x, const float* __restrict__ Wemb,
    const float* __restrict__ bemb, const float* __restrict__ pos,
    float* __restrict__ h)
{
    int idx = blockIdx.x * 256 + threadIdx.x;           // over NTOK*DIM = 4M
    if (idx >= NTOK * DIMC) return;
    int d  = idx & (DIMC - 1);
    int bt = idx >> 10;
    int t  = bt & (TLEN - 1);
    const float* xr = x + (size_t)bt * 6;
    float a = bemb[d] + pos[(size_t)t * DIMC + d];
    #pragma unroll
    for (int j = 0; j < 6; j++) a += xr[j] * Wemb[(size_t)d * 6 + j];
    h[idx] = a;
}

// ---------------------------------------------------------------------------
// f32 -> bf16 cast (weights)
// ---------------------------------------------------------------------------
__global__ __launch_bounds__(256) void cast_kernel(
    const float* __restrict__ src, __bf16* __restrict__ dst, int n)
{
    int i = blockIdx.x * 256 + threadIdx.x;
    int stride = gridDim.x * 256;
    for (; i < n; i += stride) dst[i] = (__bf16)src[i];
}

__global__ void zero1_kernel(float* p) { if (threadIdx.x == 0) p[0] = 0.f; }

// ---------------------------------------------------------------------------
// mln pass1: per-token layernorm prelude + mq, accumulate |mq| globally.
// One block (256 threads) per token row of 1024.
// ---------------------------------------------------------------------------
__global__ __launch_bounds__(256) void mln_pass1(
    const float* __restrict__ X, const float* __restrict__ pack, // pw|pb|g|b (4*1024)
    float* __restrict__ XN, float* __restrict__ MQ, float* __restrict__ accum)
{
    __shared__ float sred[256];
    int row = blockIdx.x, tid = threadIdx.x;
    const float* x = X + (size_t)row * DIMC;
    float v[4]; float s = 0.f, s2 = 0.f;
    #pragma unroll
    for (int i = 0; i < 4; i++) { v[i] = x[tid + i * 256]; s += v[i]; s2 += v[i] * v[i]; }
    sred[tid] = s; __syncthreads();
    for (int off = 128; off; off >>= 1) { if (tid < off) sred[tid] += sred[tid + off]; __syncthreads(); }
    float mu = sred[0] / (float)DIMC; __syncthreads();
    sred[tid] = s2; __syncthreads();
    for (int off = 128; off; off >>= 1) { if (tid < off) sred[tid] += sred[tid + off]; __syncthreads(); }
    float var = sred[0] / (float)DIMC - mu * mu; __syncthreads();
    float inv = rsqrtf(var + EPSC);
    float mqp = 0.f;
    #pragma unroll
    for (int i = 0; i < 4; i++) {
        int d = tid + i * 256;
        float xnv = (v[i] - mu) * inv * pack[d] + pack[DIMC + d];
        XN[(size_t)row * DIMC + d] = xnv;
        mqp += (d < TDC) ? -xnv * xnv : xnv * xnv;   // mq = sum(s^2) - sum(t^2)
    }
    sred[tid] = mqp; __syncthreads();
    for (int off = 128; off; off >>= 1) { if (tid < off) sred[tid] += sred[tid + off]; __syncthreads(); }
    if (tid == 0) { MQ[row] = sred[0]; atomicAdd(accum, fabsf(sred[0])); }
}

// mln pass2: apply per-token mq scaling with global e; emit bf16.
__global__ __launch_bounds__(256) void mln_pass2(
    const float* __restrict__ XN, const float* __restrict__ pack,
    const float* __restrict__ MQ, const float* __restrict__ accum,
    __bf16* __restrict__ OUT)
{
    int row = blockIdx.x, tid = threadIdx.x;
    float e  = fmaxf(0.01f * accum[0] / (float)NTOK, EPSC);
    float den = 1.f / (sqrtf(fabsf(MQ[row]) + e) + e);
    #pragma unroll
    for (int i = 0; i < 4; i++) {
        int d = tid + i * 256;
        float xnv = XN[(size_t)row * DIMC + d];
        OUT[(size_t)row * DIMC + d] = (__bf16)(pack[2 * DIMC + d] * (xnv * den) + pack[3 * DIMC + d]);
    }
}

// ---------------------------------------------------------------------------
// Tiled WMMA GEMM:  C(MxN) = A(MxK) @ W(NxK)^T,  A/W bf16 row-major.
// Block 256 thr = 8 waves (4x2), BM=128 BN=64 BK=32; wave tile 32x32 (2x2 WMMA).
// Tiles staged via CDNA5 GLOBAL_LOAD_ASYNC_TO_LDS_B128 with double buffering.
// EPI: 0 = write bf16       (Cb[row*ldc + colOff + col])
//      1 = write bf16 transposed per batch (Vt[b][col][t], hardcoded T=D=1024)
//      2 = Cf[row*ldc+col] += val + bias[col]   (residual f32)
//      3 = Cb[row*ldc+col] = bf16(gelu(val + bias[col]))  (exact erf gelu)
// ---------------------------------------------------------------------------
#define BM 128
#define BN 64
#define BK 32

template <int EPI>
__global__ __launch_bounds__(256) void gemm_bf16(
    const __bf16* __restrict__ A, int lda,
    const __bf16* __restrict__ W, int ldw,
    int K,
    float* __restrict__ Cf, __bf16* __restrict__ Cb, int ldc, int colOff,
    const float* __restrict__ bias)
{
    __shared__ __bf16 sA[2][BM * BK];
    __shared__ __bf16 sB[2][BN * BK];
    int tid  = threadIdx.x;
    int lane = tid & 31, wave = tid >> 5;
    int wm = wave & 3, wn = wave >> 2;
    int bm = blockIdx.x * BM, bn = blockIdx.y * BN;

    // Per-thread staging coordinates: A tile = 512 16B chunks (2/thread),
    // B tile = 256 16B chunks (1/thread).
    int arow0 = tid >> 2;                 // rows 0..63
    int arow1 = (tid + 256) >> 2;         // rows 64..127
    int acc0  = (tid & 3) * 8;
    int brow  = tid >> 2;
    int bcc   = (tid & 3) * 8;

    auto issue = [&](int buf, int k0) {
        async_copy16((unsigned)(uintptr_t)&sA[buf][arow0 * BK + acc0],
                     &A[(size_t)(bm + arow0) * lda + k0 + acc0]);
        async_copy16((unsigned)(uintptr_t)&sA[buf][arow1 * BK + acc0],
                     &A[(size_t)(bm + arow1) * lda + k0 + acc0]);
        async_copy16((unsigned)(uintptr_t)&sB[buf][brow * BK + bcc],
                     &W[(size_t)(bn + brow) * ldw + k0 + bcc]);
    };

    v8f z8 = {0.f,0.f,0.f,0.f,0.f,0.f,0.f,0.f};
    v8f acc[2][2] = {{z8, z8}, {z8, z8}};

    issue(0, 0);
    int nk = K / BK;
    for (int kk = 0; kk < nk; kk++) {
        int cur = kk & 1;
        wait_asynccnt0();        // this wave's copies into sA/sB[cur] done
        __syncthreads();         // all waves' copies visible; prev reads retired

        if (kk + 1 < nk) issue(cur ^ 1, (kk + 1) * BK);

        v16bf aF[2], bF[2];
        #pragma unroll
        for (int t = 0; t < 2; t++) aF[t] = load_frag(&sA[cur][(wm * 32 + t * 16) * BK], BK, lane);
        #pragma unroll
        for (int t = 0; t < 2; t++) bF[t] = load_frag(&sB[cur][(wn * 32 + t * 16) * BK], BK, lane);
        #pragma unroll
        for (int i = 0; i < 2; i++)
            #pragma unroll
            for (int j = 0; j < 2; j++)
                acc[i][j] = wmma_bf16(aF[i], bF[j], acc[i][j]);
        __syncthreads();         // reads of buf[cur] done before it is refilled
    }

    // Epilogue. C layout: VGPR r -> M=r (lanes 0-15) / M=r+8 (lanes 16-31), N=lane&15.
    int hl = (lane >> 4) & 1;
    int nn = lane & 15;
    #pragma unroll
    for (int i = 0; i < 2; i++) {
        #pragma unroll
        for (int j = 0; j < 2; j++) {
            int col = bn + wn * 32 + j * 16 + nn;
            int mrb = bm + wm * 32 + i * 16 + hl * 8;
            #pragma unroll
            for (int r = 0; r < 8; r++) {
                float val = acc[i][j][r];
                int row = mrb + r;
                if constexpr (EPI == 0) {
                    Cb[(size_t)row * ldc + colOff + col] = (__bf16)val;
                } else if constexpr (EPI == 1) {
                    int bi = row >> 10, t = row & 1023;
                    Cb[(((size_t)bi << 10) + col) * 1024 + t] = (__bf16)val;
                } else if constexpr (EPI == 2) {
                    Cf[(size_t)row * ldc + col] += val + bias[col];
                } else {
                    float xx = val + bias[col];
                    float g  = 0.5f * xx * (1.f + erff(xx * 0.70710678118f));
                    Cb[(size_t)row * ldc + col] = (__bf16)g;
                }
            }
        }
    }
}

// ---------------------------------------------------------------------------
// Fused attention: one block per (b, head, 16-query tile). 256 thr = 8 waves.
// Step1: S = scale * Q_blk @ K_h^T (WMMA, all 8 waves, 128 keys each)
// Step2: row softmax of S in LDS -> P (bf16)
// Step3: O = P @ V_h using transposed V (Vt: [b][d][t]) (WMMA, waves 0-3)
// ---------------------------------------------------------------------------
#define ATTN_LDS_BYTES (16*1024*4 + 16*1024*2 + 256*4)   // S + P + red

__global__ __launch_bounds__(256) void attn_kernel(
    const __bf16* __restrict__ Qb, const __bf16* __restrict__ Kb,
    const __bf16* __restrict__ Vt, __bf16* __restrict__ Ob,
    const float* __restrict__ w_sigma, int layer)
{
    extern __shared__ char smem[];
    float*  S   = (float*)smem;                         // 16 x 1024 f32
    __bf16* P   = (__bf16*)(smem + 16 * 1024 * 4);      // 16 x 1024 bf16
    float*  red = (float*)(smem + 16 * 1024 * 4 + 16 * 1024 * 2); // 256 f32

    int blk = blockIdx.x;               // B*NH*(T/16) = 4096
    int qt  = blk & 63;
    int hh  = (blk >> 6) & 15;
    int b   = blk >> 10;
    int tid = threadIdx.x, lane = tid & 31, wave = tid >> 5;

    size_t qbase = ((size_t)(b * TLEN + qt * 16)) * DIMC + hh * HDIM;
    size_t kbase = ((size_t)(b * TLEN)) * DIMC + hh * HDIM;

    float sig   = 1.f / (1.f + __expf(-w_sigma[layer]));
    float scale = (hh < NTH ? -sig : 1.f) * 0.125f;     // HD^-0.5 = 1/8

    // ---- Step 1: scores -------------------------------------------------
    v16bf qF0 = load_frag(Qb + qbase,      DIMC, lane);
    v16bf qF1 = load_frag(Qb + qbase + 32, DIMC, lane);
    int hl = (lane >> 4) & 1;
    int nn = lane & 15;
    for (int nt = 0; nt < 8; nt++) {
        int keyt = wave * 8 + nt;                       // 64 key tiles of 16
        v8f acc = {0.f,0.f,0.f,0.f,0.f,0.f,0.f,0.f};
        v16bf k0 = load_frag(Kb + kbase + (size_t)keyt * 16 * DIMC,      DIMC, lane);
        v16bf k1 = load_frag(Kb + kbase + (size_t)keyt * 16 * DIMC + 32, DIMC, lane);
        acc = wmma_bf16(qF0, k0, acc);
        acc = wmma_bf16(qF1, k1, acc);
        int n = keyt * 16 + nn;
        #pragma unroll
        for (int r = 0; r < 8; r++) S[(hl * 8 + r) * 1024 + n] = acc[r] * scale;
    }
    __syncthreads();

    // ---- Step 2: softmax ------------------------------------------------
    int row = tid >> 4, sub = tid & 15;                 // 16 threads per row
    float mx = -3.0e38f;
    for (int c = sub; c < 1024; c += 16) mx = fmaxf(mx, S[row * 1024 + c]);
    red[row * 16 + sub] = mx; __syncthreads();
    for (int off = 8; off; off >>= 1) {
        if (sub < off) red[row * 16 + sub] = fmaxf(red[row * 16 + sub], red[row * 16 + sub + off]);
        __syncthreads();
    }
    mx = red[row * 16]; __syncthreads();
    float sum = 0.f;
    for (int c = sub; c < 1024; c += 16) {
        float e = __expf(S[row * 1024 + c] - mx);
        S[row * 1024 + c] = e;
        sum += e;
    }
    red[row * 16 + sub] = sum; __syncthreads();
    for (int off = 8; off; off >>= 1) {
        if (sub < off) red[row * 16 + sub] += red[row * 16 + sub + off];
        __syncthreads();
    }
    float inv = 1.f / red[row * 16];
    for (int c = sub; c < 1024; c += 16)
        P[row * 1024 + c] = (__bf16)(S[row * 1024 + c] * inv);
    __syncthreads();

    // ---- Step 3: O = P @ V_h (waves 0..3, one 16-wide d-tile each) ------
    if (wave < 4) {
        v8f acc = {0.f,0.f,0.f,0.f,0.f,0.f,0.f,0.f};
        const __bf16* vb = Vt + ((size_t)(b * DIMC) + hh * HDIM + wave * 16) * (size_t)TLEN;
        for (int kt = 0; kt < TLEN; kt += 32) {
            v16bf pF = load_frag(P + kt, 1024, lane);
            v16bf vF = load_frag(vb + kt, TLEN, lane);
            acc = wmma_bf16(pF, vF, acc);
        }
        int d = hh * HDIM + wave * 16 + nn;
        #pragma unroll
        for (int r = 0; r < 8; r++) {
            size_t orow = (size_t)(b * TLEN + qt * 16 + hl * 8 + r);
            Ob[orow * DIMC + d] = (__bf16)acc[r];
        }
    }
}

// ---------------------------------------------------------------------------
// Head: out(B,120) = xb_last(B,1024) @ Wh^T + bh   (tiny; plain VALU)
// ---------------------------------------------------------------------------
__global__ __launch_bounds__(128) void head_kernel(
    const __bf16* __restrict__ XB, const float* __restrict__ Wh,
    const float* __restrict__ bh, float* __restrict__ out)
{
    int b = blockIdx.x, t = threadIdx.x;
    if (t < 120) {
        const __bf16* x = XB + ((size_t)(b * TLEN) + (TLEN - 1)) * DIMC;
        const float*  w = Wh + (size_t)t * DIMC;
        float acc = bh[t];
        for (int d = 0; d < DIMC; d++) acc += (float)x[d] * w[d];
        out[b * 120 + t] = acc;
    }
}

// ---------------------------------------------------------------------------
// Driver
// ---------------------------------------------------------------------------
extern "C" void kernel_launch(void* const* d_in, const int* in_sizes, int n_in,
                              void* d_out, int out_size, void* d_ws, size_t ws_size,
                              hipStream_t stream)
{
    const float* in_x    = (const float*)d_in[0];
    const float* in_Wemb = (const float*)d_in[1];
    const float* in_bemb = (const float*)d_in[2];
    const float* in_pos  = (const float*)d_in[3];
    const float* in_n1   = (const float*)d_in[4];
    const float* in_n2   = (const float*)d_in[5];
    const float* in_nf   = (const float*)d_in[6];
    const float* in_Wqt  = (const float*)d_in[7];
    const float* in_Wkt  = (const float*)d_in[8];
    const float* in_Wqs  = (const float*)d_in[9];
    const float* in_Wks  = (const float*)d_in[10];
    const float* in_Wv   = (const float*)d_in[11];
    const float* in_Wout = (const float*)d_in[12];
    const float* in_bout = (const float*)d_in[13];
    const float* in_wsig = (const float*)d_in[14];
    const float* in_W1   = (const float*)d_in[15];
    const float* in_b1   = (const float*)d_in[16];
    const float* in_W2   = (const float*)d_in[17];
    const float* in_b2   = (const float*)d_in[18];
    const float* in_Wh   = (const float*)d_in[19];
    const float* in_bh   = (const float*)d_in[20];
    float* out = (float*)d_out;

    // Workspace layout (~129 MB)
    const size_t MB = 1ull << 20;
    char* W = (char*)d_ws;
    float*  h     = (float*) (W + 0);          // 16 MB  (NTOK x DIM f32)
    float*  xn    = (float*) (W + 16 * MB);    // 16 MB  (mln temp)
    __bf16* xb    = (__bf16*)(W + 32 * MB);    //  8 MB  (bf16 activations)
    __bf16* qb    = (__bf16*)(W + 40 * MB);    //  8 MB
    __bf16* kb    = (__bf16*)(W + 48 * MB);    //  8 MB
    __bf16* vt    = (__bf16*)(W + 56 * MB);    //  8 MB  (V transposed: [b][d][t])
    __bf16* ob    = (__bf16*)(W + 64 * MB);    //  8 MB
    __bf16* gb    = (__bf16*)(W + 72 * MB);    // 32 MB  (NTOK x 4DIM bf16)
    __bf16* wc    = (__bf16*)(W + 104 * MB);   // 24 MB  (per-layer bf16 weights)
    float*  mqb   = (float*) (W + 128 * MB);   // 16 KB
    float*  accum = (float*) (W + 128 * MB + 65536);

    // per-layer bf16 weight slots (element offsets into wc)
    __bf16* wqt = wc;
    __bf16* wkt = wc + 262144;
    __bf16* wqs = wc + 524288;
    __bf16* wks = wc + 1310720;
    __bf16* wv  = wc + 2097152;
    __bf16* wo  = wc + 3145728;
    __bf16* w1  = wc + 4194304;
    __bf16* w2  = wc + 8388608;

    // 1) Embedding
    embed_kernel<<<(NTOK * DIMC) / 256, 256, 0, stream>>>(in_x, in_Wemb, in_bemb, in_pos, h);

    for (int i = 0; i < LAY; i++) {
        const float* n1p = in_n1 + (size_t)i * 4 * DIMC;
        const float* n2p = in_n2 + (size_t)i * 4 * DIMC;

        // ---- mln1 -> xb (bf16) ----
        zero1_kernel<<<1, 32, 0, stream>>>(accum);
        mln_pass1<<<NTOK, 256, 0, stream>>>(h, n1p, xn, mqb, accum);
        mln_pass2<<<NTOK, 256, 0, stream>>>(xn, n1p, mqb, accum, xb);

        // ---- cast this layer's weights to bf16 ----
        cast_kernel<<<1024, 256, 0, stream>>>(in_Wqt + (size_t)i * 262144,  wqt, 262144);
        cast_kernel<<<1024, 256, 0, stream>>>(in_Wkt + (size_t)i * 262144,  wkt, 262144);
        cast_kernel<<<1024, 256, 0, stream>>>(in_Wqs + (size_t)i * 786432,  wqs, 786432);
        cast_kernel<<<1024, 256, 0, stream>>>(in_Wks + (size_t)i * 786432,  wks, 786432);
        cast_kernel<<<2048, 256, 0, stream>>>(in_Wv  + (size_t)i * 1048576, wv,  1048576);
        cast_kernel<<<2048, 256, 0, stream>>>(in_Wout+ (size_t)i * 1048576, wo,  1048576);
        cast_kernel<<<4096, 256, 0, stream>>>(in_W1  + (size_t)i * 4194304, w1,  4194304);
        cast_kernel<<<4096, 256, 0, stream>>>(in_W2  + (size_t)i * 4194304, w2,  4194304);

        // ---- projections (WMMA GEMMs) ----
        gemm_bf16<0><<<dim3(NTOK/BM,  256/BN), 256, 0, stream>>>(xb, DIMC, wqt, DIMC, DIMC, nullptr, qb, DIMC,   0, nullptr);
        gemm_bf16<0><<<dim3(NTOK/BM,  768/BN), 256, 0, stream>>>(xb, DIMC, wqs, DIMC, DIMC, nullptr, qb, DIMC, TDC, nullptr);
        gemm_bf16<0><<<dim3(NTOK/BM,  256/BN), 256, 0, stream>>>(xb, DIMC, wkt, DIMC, DIMC, nullptr, kb, DIMC,   0, nullptr);
        gemm_bf16<0><<<dim3(NTOK/BM,  768/BN), 256, 0, stream>>>(xb, DIMC, wks, DIMC, DIMC, nullptr, kb, DIMC, TDC, nullptr);
        gemm_bf16<1><<<dim3(NTOK/BM, DIMC/BN), 256, 0, stream>>>(xb, DIMC, wv,  DIMC, DIMC, nullptr, vt, DIMC,   0, nullptr);

        // ---- fused attention ----
        attn_kernel<<<BATCH * NHEAD * (TLEN / 16), 256, ATTN_LDS_BYTES, stream>>>(
            qb, kb, vt, ob, in_wsig, i);

        // ---- h += attn_out @ Wout^T + bout ----
        gemm_bf16<2><<<dim3(NTOK/BM, DIMC/BN), 256, 0, stream>>>(ob, DIMC, wo, DIMC, DIMC, h, nullptr, DIMC, 0, in_bout + (size_t)i * DIMC);

        // ---- mln2 -> xb ----
        zero1_kernel<<<1, 32, 0, stream>>>(accum);
        mln_pass1<<<NTOK, 256, 0, stream>>>(h, n2p, xn, mqb, accum);
        mln_pass2<<<NTOK, 256, 0, stream>>>(xn, n2p, mqb, accum, xb);

        // ---- MLP ----
        gemm_bf16<3><<<dim3(NTOK/BM, (4*DIMC)/BN), 256, 0, stream>>>(xb, DIMC, w1, DIMC, DIMC, nullptr, gb, 4*DIMC, 0, in_b1 + (size_t)i * 4 * DIMC);
        gemm_bf16<2><<<dim3(NTOK/BM,     DIMC/BN), 256, 0, stream>>>(gb, 4*DIMC, w2, 4*DIMC, 4*DIMC, h, nullptr, DIMC, 0, in_b2 + (size_t)i * DIMC);
    }

    // final mln (e depends on ALL tokens, so run over everything)
    zero1_kernel<<<1, 32, 0, stream>>>(accum);
    mln_pass1<<<NTOK, 256, 0, stream>>>(h, in_nf, xn, mqb, accum);
    mln_pass2<<<NTOK, 256, 0, stream>>>(xn, in_nf, mqb, accum, xb);

    // head on last token of each batch
    head_kernel<<<BATCH, 128, 0, stream>>>(xb, in_Wh, in_bh, out);
}
